// CustomLSTMCell_33114197852278
// MI455X (gfx1250) — matrix operand order repaired and checked by reading, not compiled, not run on hardware
//
#include <hip/hip_runtime.h>

#define B_SZ 8192
#define D_SZ 1024
#define H_SZ 1024
#define G_SZ 4096      // 4*H
#define K_SZ 2048      // D + H

#define BM 128
#define BN 128
#define BK 64
#define LDA 72         // padded ushorts per LDS row (144B stride)
#define NSTAGE (K_SZ / BK)

typedef __attribute__((ext_vector_type(4)))  int    v4i;
typedef __attribute__((ext_vector_type(16))) __bf16 v16bf;
typedef __attribute__((ext_vector_type(8)))  float  v8f;

#if defined(__gfx1250__) && __has_builtin(__builtin_amdgcn_global_load_async_to_lds_b128)
#define USE_ASYNC 1
#else
#define USE_ASYNC 0
#endif

#if USE_ASYNC
// Builtin signature: (v4i addrspace(1)*, v4i addrspace(3)*, imm offset, imm cpol).
// IOFFSET is added to BOTH the global and the LDS address (ISA async-to-LDS ops),
// so one base pair + immediate offsets moves a contiguous 64B run.
#define ASYNC16(gp, lp, off)                                               \
  __builtin_amdgcn_global_load_async_to_lds_b128(                          \
      (__attribute__((address_space(1))) v4i*)(void*)(gp),                 \
      (__attribute__((address_space(3))) v4i*)(void*)(lp), (off), 0)
#endif

__device__ __forceinline__ void wait_async_all() {
#if defined(__gfx1250__)
#if __has_builtin(__builtin_amdgcn_s_wait_asynccnt)
  __builtin_amdgcn_s_wait_asynccnt(0);
#else
  asm volatile("s_wait_asynccnt 0" ::: "memory");
#endif
#endif
}

__device__ __forceinline__ unsigned int f2bf(float f) {
  unsigned int u = __float_as_uint(f);
  u += 0x7FFFu + ((u >> 16) & 1u);   // round-to-nearest-even
  return u >> 16;
}

__device__ __forceinline__ float sig_(float x)  { return 1.0f / (1.0f + __expf(-x)); }
__device__ __forceinline__ float tanh_(float x) { return 1.0f - 2.0f / (__expf(2.0f * x) + 1.0f); }

// ---- pre-pass: f32 -> bf16, concatenated [e_t | h_prev] -> Abig[8192][2048]
__global__ __launch_bounds__(256)
void cvt_act(const float* __restrict__ e_t, const float* __restrict__ h_prev,
             unsigned short* __restrict__ Abig) {
  size_t idx  = (size_t)blockIdx.x * 256 + threadIdx.x;
  size_t base = idx * 8;
  size_t b = base / K_SZ;
  int    k = (int)(base % K_SZ);
  const float* src = (k < D_SZ) ? (e_t + b * D_SZ + k) : (h_prev + b * H_SZ + (k - D_SZ));
  float4 f0 = *(const float4*)(src);
  float4 f1 = *(const float4*)(src + 4);
  uint4 q;
  q.x = f2bf(f0.x) | (f2bf(f0.y) << 16);
  q.y = f2bf(f0.z) | (f2bf(f0.w) << 16);
  q.z = f2bf(f1.x) | (f2bf(f1.y) << 16);
  q.w = f2bf(f1.z) | (f2bf(f1.w) << 16);
  *(uint4*)(Abig + base) = q;
}

// ---- pre-pass: f32 -> bf16, gate-interleaved [W_x | W_h] -> Wbig[4096][2048]
// physical row r = j*4 + gate  ->  source row g = gate*H + j
__global__ __launch_bounds__(256)
void cvt_w(const float* __restrict__ W_x, const float* __restrict__ W_h,
           unsigned short* __restrict__ Wbig) {
  size_t idx  = (size_t)blockIdx.x * 256 + threadIdx.x;
  size_t base = idx * 8;
  int r = (int)(base / K_SZ);
  int k = (int)(base % K_SZ);
  int g = (r & 3) * H_SZ + (r >> 2);
  const float* src = (k < D_SZ) ? (W_x + (size_t)g * D_SZ + k)
                                : (W_h + (size_t)g * H_SZ + (k - D_SZ));
  float4 f0 = *(const float4*)(src);
  float4 f1 = *(const float4*)(src + 4);
  uint4 q;
  q.x = f2bf(f0.x) | (f2bf(f0.y) << 16);
  q.y = f2bf(f0.z) | (f2bf(f0.w) << 16);
  q.z = f2bf(f1.x) | (f2bf(f1.y) << 16);
  q.w = f2bf(f1.z) | (f2bf(f1.w) << 16);
  *(uint4*)(Wbig + base) = q;
}

// ---- pre-pass: summed bias in interleaved order
__global__ __launch_bounds__(256)
void cvt_bias(const float* __restrict__ b_x, const float* __restrict__ b_h,
              const float* __restrict__ b_ex, float* __restrict__ biasI) {
  int r = blockIdx.x * 256 + threadIdx.x;
  int g = (r & 3) * H_SZ + (r >> 2);
  biasI[r] = b_x[g] + b_h[g] + b_ex[g];
}

// ---- main fused kernel: z = Abig @ Wbig^T + bias, then LSTM combine
__global__ __launch_bounds__(256)
void lstm_wmma(const unsigned short* __restrict__ Abig,
               const unsigned short* __restrict__ Wbig,
               const float* __restrict__ biasI,
               const float* __restrict__ c_prev,
               float* __restrict__ h_out,
               float* __restrict__ c_out) {
  __shared__ __align__(16) unsigned char smem[2 * (BM + BN) * LDA * 2]; // 73728 B
  unsigned short* sA = (unsigned short*)smem;     // [2][BM*LDA]
  unsigned short* sB = sA + 2 * BM * LDA;         // [2][BN*LDA]

  const int tid  = threadIdx.x;
  const int lane = tid & 31;
  const int wave = tid >> 5;
  const int l16  = lane & 15;
  const int lh   = lane >> 4;          // 0/1 : lane half
  const int wm   = (wave >> 2) * 64;   // wave M offset: 0 / 64
  const int wn   = (wave & 3) * 32;    // wave N offset: 0..96

  const int m0 = blockIdx.y * BM;
  const int n0 = blockIdx.x * BN;

  // tile staging: 2 threads per row, each moves 32 contiguous ushorts (64B)
  const int row = tid >> 1;
  const int par = tid & 1;

  const unsigned short* gA = Abig + (size_t)(m0 + row) * K_SZ + par * 32;
  const unsigned short* gB = Wbig + (size_t)(n0 + row) * K_SZ + par * 32;
  unsigned short* lA = sA + row * LDA + par * 32;   // buffer-0 LDS dest
  unsigned short* lB = sB + row * LDA + par * 32;

#if USE_ASYNC
  // stage 0: async DMA into buffer 0
  ASYNC16(gA, lA, 0);  ASYNC16(gA, lA, 16);  ASYNC16(gA, lA, 32);  ASYNC16(gA, lA, 48);
  ASYNC16(gB, lB, 0);  ASYNC16(gB, lB, 16);  ASYNC16(gB, lB, 32);  ASYNC16(gB, lB, 48);
  wait_async_all();
  __syncthreads();
#else
  uint4 ra[4], rb[4];
#pragma unroll
  for (int i = 0; i < 4; ++i) {
    ra[i] = *(const uint4*)(gA + i * 8);
    rb[i] = *(const uint4*)(gB + i * 8);
  }
#pragma unroll
  for (int i = 0; i < 4; ++i) {
    *(uint4*)(lA + i * 8) = ra[i];
    *(uint4*)(lB + i * 8) = rb[i];
  }
  __syncthreads();
#endif

  const v8f vzero = {0.f, 0.f, 0.f, 0.f, 0.f, 0.f, 0.f, 0.f};
  v8f acc[4][2];
#pragma unroll
  for (int s = 0; s < 4; ++s)
#pragma unroll
    for (int t = 0; t < 2; ++t)
      acc[s][t] = vzero;

  union Frag { uint4 q[2]; v16bf v; };

  // single loop body: keeps the 8 accumulator tuples pinned across iterations
  // (unrolling caused cross-body accumulator copies + WMMA->VALU hazard NOPs)
#pragma clang loop unroll(disable)
  for (int kt = 0; kt < NSTAGE; ++kt) {
    const int buf = kt & 1;

#if USE_ASYNC
    // kick off async DMA for the next stage into the other buffer
    if (kt + 1 < NSTAGE) {
      const unsigned short* pA = gA + (size_t)(kt + 1) * BK;
      const unsigned short* pB = gB + (size_t)(kt + 1) * BK;
      unsigned short* dA = lA + (buf ^ 1) * (BM * LDA);
      unsigned short* dB = lB + (buf ^ 1) * (BN * LDA);
      ASYNC16(pA, dA, 0);  ASYNC16(pA, dA, 16);  ASYNC16(pA, dA, 32);  ASYNC16(pA, dA, 48);
      ASYNC16(pB, dB, 0);  ASYNC16(pB, dB, 16);  ASYNC16(pB, dB, 32);  ASYNC16(pB, dB, 48);
    }
#else
    if (kt + 1 < NSTAGE) {
      const unsigned short* pA = gA + (size_t)(kt + 1) * BK;
      const unsigned short* pB = gB + (size_t)(kt + 1) * BK;
#pragma unroll
      for (int i = 0; i < 4; ++i) {
        ra[i] = *(const uint4*)(pA + i * 8);
        rb[i] = *(const uint4*)(pB + i * 8);
      }
    }
#endif

    const unsigned short* bA = sA + buf * (BM * LDA);
    const unsigned short* bB = sB + buf * (BN * LDA);
#pragma unroll
    for (int kk = 0; kk < BK; kk += 32) {
      Frag fa[4], fb[2];
      // A 16x32 bf16 frag: lanes 0-15 rows M, K0-7 + K16-23; lanes 16-31 K8-15 + K24-31
#pragma unroll
      for (int s = 0; s < 4; ++s) {
        const unsigned short* p = bA + (wm + s * 16 + l16) * LDA + kk + lh * 8;
        fa[s].q[0] = *(const uint4*)(p);
        fa[s].q[1] = *(const uint4*)(p + 16);
      }
      // B 32x16 bf16 frag: lanes 0-15 col N, K0-7 + K16-23; lanes 16-31 K8-15 + K24-31
#pragma unroll
      for (int t = 0; t < 2; ++t) {
        const unsigned short* p = bB + (wn + t * 16 + l16) * LDA + kk + lh * 8;
        fb[t].q[0] = *(const uint4*)(p);
        fb[t].q[1] = *(const uint4*)(p + 16);
      }
#pragma unroll
      for (int s = 0; s < 4; ++s)
#pragma unroll
        for (int t = 0; t < 2; ++t)
          acc[s][t] = __builtin_amdgcn_wmma_f32_16x16x32_bf16(
              false, fa[s].v, false, fb[t].v, (short)0, acc[s][t], false, false);
    }

#if USE_ASYNC
    if (kt + 1 < NSTAGE) wait_async_all();  // next-stage DMA landed in LDS
    __syncthreads();                        // visible to all waves; fences this stage's ds reads
#else
    if (kt + 1 < NSTAGE) {
      unsigned short* dA = lA + (buf ^ 1) * (BM * LDA);
      unsigned short* dB = lB + (buf ^ 1) * (BN * LDA);
#pragma unroll
      for (int i = 0; i < 4; ++i) {
        *(uint4*)(dA + i * 8) = ra[i];
        *(uint4*)(dB + i * 8) = rb[i];
      }
    }
    __syncthreads();
#endif
  }

  // ---- epilogue: dump z tile (+bias) to LDS (reusing staging mem), then fuse LSTM
  float* zs = (float*)smem;  // [BM][132]
  float bv[2];
#pragma unroll
  for (int t = 0; t < 2; ++t) bv[t] = biasI[n0 + wn + t * 16 + l16];
#pragma unroll
  for (int s = 0; s < 4; ++s) {
#pragma unroll
    for (int t = 0; t < 2; ++t) {
      const int mb = wm + s * 16 + lh * 8;   // C/D layout: lanes 16-31 hold M = e+8
      const int nn = wn + t * 16 + l16;
#pragma unroll
      for (int e = 0; e < 8; ++e)
        zs[(mb + e) * 132 + nn] = acc[s][t][e] + bv[t];
    }
  }
  __syncthreads();

  const int jbase = n0 >> 2;
#pragma unroll
  for (int i = 0; i < 16; ++i) {
    const int o  = tid + 256 * i;          // 128 rows x 32 j-cols = 4096 outputs
    const int ml = o >> 5;
    const int jl = o & 31;
    float4 v = *(const float4*)(zs + ml * 132 + jl * 4); // (zf, zi, zo, zc)
    const size_t idx = (size_t)(m0 + ml) * H_SZ + (jbase + jl);
    const float cp = c_prev[idx];
    const float gf = sig_(v.x);
    const float pi = sig_(v.y);
    const float qo = sig_(v.z);
    const float mh = tanh_(v.w);
    const float ct = gf * cp + pi * mh;
    const float ht = qo * tanh_(ct);
    c_out[idx] = ct;
    h_out[idx] = ht;
  }
}

extern "C" void kernel_launch(void* const* d_in, const int* in_sizes, int n_in,
                              void* d_out, int out_size, void* d_ws, size_t ws_size,
                              hipStream_t stream) {
  const float* e_t    = (const float*)d_in[0];
  const float* h_prev = (const float*)d_in[1];
  const float* c_prev = (const float*)d_in[2];
  const float* W_x    = (const float*)d_in[3];
  const float* b_x    = (const float*)d_in[4];
  const float* W_h    = (const float*)d_in[5];
  const float* b_h    = (const float*)d_in[6];
  const float* b_ex   = (const float*)d_in[7];

  unsigned char* ws = (unsigned char*)d_ws;
  unsigned short* Abig = (unsigned short*)ws;                                  // 33,554,432 B
  unsigned short* Wbig = (unsigned short*)(ws + (size_t)B_SZ * K_SZ * 2);      // 16,777,216 B
  float* biasI = (float*)(ws + (size_t)B_SZ * K_SZ * 2 + (size_t)G_SZ * K_SZ * 2); // 16 KB

  float* h_out = (float*)d_out;
  float* c_out = h_out + (size_t)B_SZ * H_SZ;

  cvt_act <<<(B_SZ * (K_SZ / 8)) / 256, 256, 0, stream>>>(e_t, h_prev, Abig);
  cvt_w   <<<(G_SZ * (K_SZ / 8)) / 256, 256, 0, stream>>>(W_x, W_h, Wbig);
  cvt_bias<<<G_SZ / 256, 256, 0, stream>>>(b_x, b_h, b_ex, biasI);

  dim3 grid(G_SZ / BN, B_SZ / BM);   // 32 x 64 = 2048 workgroups
  lstm_wmma<<<grid, 256, 0, stream>>>(Abig, Wbig, biasI, c_prev, h_out, c_out);
}
